// DB_30571577212981
// MI455X (gfx1250) — matrix operand order repaired and checked by reading, not compiled, not run on hardware
//
#include <hip/hip_runtime.h>
#include <hip/hip_bf16.h>

typedef __attribute__((ext_vector_type(16))) _Float16 v16h;
typedef __attribute__((ext_vector_type(8)))  float    v8f;

#define NIMG 2
#define C_   64
#define H_   128
#define W_   128
#define HW_  16384
#define OC_  32
#define DG_  8
#define CG_  8
#define LRS  0.2f

__device__ __forceinline__ float sigm(float x) { return 1.f / (1.f + expf(-x)); }

// ---------------- weight convert fp32 -> f16, pad rows to Mpad ----------------
__global__ void k_wcvt(const float* __restrict__ w, _Float16* __restrict__ o,
                       int M, int Mpad, int K) {
  int idx = blockIdx.x * blockDim.x + threadIdx.x;
  if (idx >= Mpad * K) return;
  int m = idx / K;
  o[idx] = (m < M) ? (_Float16)w[idx] : (_Float16)0.f;
}

// ---------------- im2col (k=3, pad=1, dil=1) -> [img][HW][C*9] f16 ------------
__global__ void k_im2col(const float* __restrict__ x, _Float16* __restrict__ col,
                         int C) {
  int idx = blockIdx.x * blockDim.x + threadIdx.x;
  int total = NIMG * HW_ * C;
  if (idx >= total) return;
  int c = idx % C;
  int p = (idx / C) % HW_;
  int img = idx / (C * HW_);
  int h = p >> 7, w = p & 127;
  const float* xc = x + ((size_t)img * C + c) * HW_;
  _Float16* o = col + ((size_t)img * HW_ + p) * (size_t)(C * 9) + (size_t)c * 9;
#pragma unroll
  for (int ky = 0; ky < 3; ++ky)
#pragma unroll
    for (int kx = 0; kx < 3; ++kx) {
      int yy = h + ky - 1, xx = w + kx - 1;
      float v = (yy >= 0 && yy < H_ && xx >= 0 && xx < W_) ? xc[yy * W_ + xx] : 0.f;
      o[ky * 3 + kx] = (_Float16)v;
    }
}

// ---------------- generic WMMA GEMM:  D[M,HW] = A[Mpad,K] x Bt[HW,K]^T --------
// A row-major [Mpad,K] f16 ; Bt pixel-major [img][HW][K] f16 (K contiguous).
// One wave computes a 16(M) x 64(N) strip: 4 accumulators share one A fragment,
// 4x v_wmma_f32_16x16x32_f16 per k-step.
__global__ void k_gemm(const _Float16* __restrict__ A,
                       const _Float16* __restrict__ Bt,
                       const float* __restrict__ bias,
                       float* __restrict__ Cf, _Float16* __restrict__ Ch,
                       const float* __restrict__ resid,
                       int M, int K, int act) {
  int lane = threadIdx.x & 31;
  int wv   = threadIdx.x >> 5;
  int half = lane >> 4;
  int r    = lane & 15;
  int p0   = (blockIdx.x * 4 + wv) * 64;   // 64 pixels per wave
  int m0   = blockIdx.y * 16;
  int img  = blockIdx.z;

  const unsigned int* Arow = (const unsigned int*)(A + (size_t)(m0 + r) * K);
  const unsigned int* Brow = (const unsigned int*)(Bt + (size_t)img * HW_ * K +
                                                   (size_t)(p0 + r) * K);
  const int tstep = 8 * K;                  // dwords between 16-pixel tiles
  v8f acc[4] = {};
  for (int k0 = 0; k0 < K; k0 += 32) {
    union { v16h v; unsigned int u[8]; } a, b[4];
    int ka = (k0 >> 1) + half * 8;          // A: lanes<16 K 0-15 ; lanes>=16 K 16-31
#pragma unroll
    for (int t = 0; t < 4; ++t) { a.u[t] = Arow[ka + t]; a.u[4 + t] = Arow[ka + 4 + t]; }
    int kb = (k0 >> 1) + half * 8;          // B: lanes<16 K 0-15 ; lanes>=16 K 16-31
#pragma unroll
    for (int j = 0; j < 4; ++j)
#pragma unroll
      for (int t = 0; t < 8; ++t) b[j].u[t] = Brow[kb + j * tstep + t];
    if (k0 + 32 < K) {                      // speculative prefetch of next k-step
      __builtin_prefetch(Arow + ka + 16, 0, 1);
      __builtin_prefetch(Brow + kb + 16, 0, 1);
    }
#pragma unroll
    for (int j = 0; j < 4; ++j)
      acc[j] = __builtin_amdgcn_wmma_f32_16x16x32_f16(false, a.v, false, b[j].v,
                                                      (short)0, acc[j], false, false);
  }
  size_t cbase = (size_t)img * M * HW_;
#pragma unroll
  for (int i = 0; i < 8; ++i) {
    int m = m0 + half * 8 + i;
    if (m < M) {
      float bs = bias ? bias[m] : 0.f;
      size_t rb = cbase + (size_t)m * HW_ + p0 + r;
#pragma unroll
      for (int j = 0; j < 4; ++j) {
        float v = acc[j][i] + bs;
        if (act) v = (v >= 0.f) ? v : LRS * v;
        size_t o = rb + j * 16;
        if (resid) v += resid[o];
        if (Cf) Cf[o] = v;
        if (Ch) Ch[o] = (_Float16)v;
      }
    }
  }
}

// ---------------- deformable sampling -> modulated columns [img][HW][576] -----
__global__ void k_dcn_cols(const float* __restrict__ x, const float* __restrict__ om,
                           _Float16* __restrict__ col) {
  int idx = blockIdx.x * blockDim.x + threadIdx.x; // img*HW*DG
  if (idx >= NIMG * HW_ * DG_) return;
  int g = idx & 7;
  int p = (idx >> 3) & (HW_ - 1);
  int img = idx >> 17;
  int h = p >> 7, w = p & 127;
  const float* omb = om + (size_t)img * 216 * HW_;
  const float* xb  = x  + (size_t)img * C_  * HW_;
  _Float16* o = col + ((size_t)img * HW_ + p) * 576 + g * 72;
  for (int k = 0; k < 9; ++k) {
    int ky = k / 3, kx = k % 3;
    float offy = omb[(g * 9 + k) * HW_ + p];
    float offx = omb[(72 + g * 9 + k) * HW_ + p];
    float mask = sigm(omb[(144 + g * 9 + k) * HW_ + p]);
    float py = (float)(h + ky - 1) + offy;
    float px = (float)(w + kx - 1) + offx;
    float y0f = floorf(py), x0f = floorf(px);
    float ly = py - y0f, lx = px - x0f;
    int iy0 = (int)y0f, ix0 = (int)x0f;
    float wt[4]; int id[4];
#pragma unroll
    for (int j = 0; j < 4; ++j) {
      int yi = iy0 + (j >> 1), xi = ix0 + (j & 1);
      float wy = (j >> 1) ? ly : (1.f - ly);
      float wx = (j & 1) ? lx : (1.f - lx);
      bool vld = (yi >= 0) && (yi <= H_ - 1) && (xi >= 0) && (xi <= W_ - 1);
      int yc = yi < 0 ? 0 : (yi > H_ - 1 ? H_ - 1 : yi);
      int xc = xi < 0 ? 0 : (xi > W_ - 1 ? W_ - 1 : xi);
      wt[j] = vld ? wy * wx : 0.f;
      id[j] = yc * W_ + xc;
    }
#pragma unroll
    for (int c = 0; c < CG_; ++c) {
      const float* xc = xb + (size_t)(g * CG_ + c) * HW_;
      float s = wt[0] * xc[id[0]] + wt[1] * xc[id[1]] +
                wt[2] * xc[id[2]] + wt[3] * xc[id[3]];
      o[c * 9 + k] = (_Float16)(s * mask);
    }
  }
}

// ---------------- per-channel mean of y1 (for SEKG eca) -----------------------
__global__ void k_chanmean(const float* __restrict__ y1, float* __restrict__ mean) {
  __shared__ float s[256];
  int b = blockIdx.x; // img*64+c
  const float* v = y1 + (size_t)b * HW_;
  float acc = 0.f;
  for (int i = threadIdx.x; i < HW_; i += 256) acc += v[i];
  s[threadIdx.x] = acc; __syncthreads();
  for (int st = 128; st > 0; st >>= 1) {
    if (threadIdx.x < st) s[threadIdx.x] += s[threadIdx.x + st];
    __syncthreads();
  }
  if (threadIdx.x == 0) mean[b] = s[0] / (float)HW_;
}

__global__ void k_ca(const float* __restrict__ mean, const float* __restrict__ caw,
                     const float* __restrict__ cab, float* __restrict__ ca) {
  int t = threadIdx.x;
  if (t >= NIMG * C_) return;
  int n = t >> 6, c = t & 63;
  const float* m = mean + n * C_;
  float l = (c > 0) ? m[c - 1] : 0.f;
  float rr = (c < C_ - 1) ? m[c + 1] : 0.f;
  ca[t] = caw[0] * l + caw[1] * m[c] + caw[2] * rr + cab[0];
}

// ---------------- SEKG: depthwise 3x3 + channel term; write f16 B [HW][64] ----
__global__ void k_sekg(const float* __restrict__ y1, const float* __restrict__ saw,
                       const float* __restrict__ sab, const float* __restrict__ ca,
                       _Float16* __restrict__ y2h) {
  int idx = blockIdx.x * blockDim.x + threadIdx.x;
  if (idx >= NIMG * C_ * HW_) return;
  int p = idx & (HW_ - 1);
  int c = (idx >> 14) & 63;
  int n = idx >> 20;
  int h = p >> 7, w = p & 127;
  const float* src = y1 + ((size_t)n * C_ + c) * HW_;
  float acc = sab[c] + ca[n * C_ + c];
#pragma unroll
  for (int ky = 0; ky < 3; ++ky)
#pragma unroll
    for (int kx = 0; kx < 3; ++kx) {
      int yy = h + ky - 1, xx = w + kx - 1;
      if (yy >= 0 && yy < H_ && xx >= 0 && xx < W_)
        acc += saw[c * 9 + ky * 3 + kx] * src[yy * W_ + xx];
    }
  y2h[((size_t)n * HW_ + p) * C_ + c] = (_Float16)acc;
}

// ---------------- dynamic per-pixel depthwise 3x3, dilation d ------------------
__global__ void k_dyndw(const float* __restrict__ x, const _Float16* __restrict__ kern,
                        float* __restrict__ tmp, int d) {
  int idx = blockIdx.x * blockDim.x + threadIdx.x;
  if (idx >= NIMG * C_ * HW_) return;
  int p = idx & (HW_ - 1);
  int c = (idx >> 14) & 63;
  int n = idx >> 20;
  int h = p >> 7, w = p & 127;
  const float* src = x + ((size_t)n * C_ + c) * HW_;
  const _Float16* kr = kern + ((size_t)n * 576 + (size_t)c * 9) * HW_ + p;
  float acc = 0.f;
#pragma unroll
  for (int k = 0; k < 9; ++k) {
    int yy = h + (k / 3 - 1) * d, xx = w + (k % 3 - 1) * d;
    if (yy >= 0 && yy < H_ && xx >= 0 && xx < W_)
      acc += (float)kr[(size_t)k * HW_] * src[yy * W_ + xx];
  }
  tmp[idx] = acc;
}

// ---------------- attention reductions over cat = [x,x1,x3,x5] ----------------
__global__ void k_catreduce(const float* __restrict__ x,  const float* __restrict__ x1,
                            const float* __restrict__ x3, const float* __restrict__ x5,
                            float* __restrict__ rowsum, float* __restrict__ colsum,
                            float* __restrict__ chansum) {
  const float* bufs[4] = {x, x1, x3, x5};
  int b = blockIdx.x;            // n*256 + cc
  int n = b >> 8, cc = b & 255;
  const float* v = bufs[cc >> 6] + ((size_t)n * C_ + (cc & 63)) * HW_;
  int t = threadIdx.x;           // 0..127
  float cs = 0.f;
  for (int h = 0; h < H_; ++h) cs += v[h * W_ + t];
  colsum[(size_t)b * W_ + t] = cs;
  float rs = 0.f;
  for (int w = 0; w < W_; ++w) rs += v[t * W_ + w];
  rowsum[(size_t)b * H_ + t] = rs;
  __shared__ float s[128];
  s[t] = rs; __syncthreads();
  for (int st = 64; st > 0; st >>= 1) {
    if (t < st) s[t] += s[t + st];
    __syncthreads();
  }
  if (t == 0) chansum[b] = s[0];
}

__global__ void k_att(const float* __restrict__ rowsum, const float* __restrict__ colsum,
                      const float* __restrict__ chansum, float* __restrict__ shw,
                      float* __restrict__ scw, float* __restrict__ sch) {
  __shared__ float sh[256];
  int n = blockIdx.x, t = threadIdx.x;
  for (int c = t; c < 256; c += 128) {
    float s = 2.f * sigm(chansum[n * 256 + c] / (float)HW_);
    sh[c] = s;
    shw[n * 256 + c] = s;
  }
  __syncthreads();
  float a = 0.f, b = 0.f;
  for (int c = 0; c < 256; ++c) {
    a += sh[c] * rowsum[((size_t)n * 256 + c) * H_ + t];
    b += sh[c] * colsum[((size_t)n * 256 + c) * W_ + t];
  }
  scw[n * H_ + t] = 2.f * sigm(a / (256.f * 128.f));
  sch[n * W_ + t] = 2.f * sigm(b / (256.f * 128.f));
}

// ---------------- build fusion B: cat * s_hw * (1+s_cw+s_ch) -> [HW][256] f16 --
__global__ void k_fusb(const float* __restrict__ x,  const float* __restrict__ x1,
                       const float* __restrict__ x3, const float* __restrict__ x5,
                       const float* __restrict__ shw, const float* __restrict__ scw,
                       const float* __restrict__ sch, _Float16* __restrict__ col) {
  const float* bufs[4] = {x, x1, x3, x5};
  int idx = blockIdx.x * blockDim.x + threadIdx.x;
  if (idx >= NIMG * HW_) return;
  int n = idx >> 14, p = idx & (HW_ - 1);
  int h = p >> 7, w = p & 127;
  float sc = 1.f + scw[n * H_ + h] + sch[n * W_ + w];
  _Float16* o = col + ((size_t)n * HW_ + p) * 256;
  for (int cc = 0; cc < 256; ++cc) {
    const float* v = bufs[cc >> 6] + ((size_t)n * C_ + (cc & 63)) * HW_;
    o[cc] = (_Float16)(v[p] * shw[n * 256 + cc] * sc);
  }
}

// =============================== host launch ==================================
extern "C" void kernel_launch(void* const* d_in, const int* in_sizes, int n_in,
                              void* d_out, int out_size, void* d_ws, size_t ws_size,
                              hipStream_t stream) {
  (void)in_sizes; (void)n_in; (void)out_size; (void)ws_size;
  const float* x        = (const float*)d_in[0];
  const float* offset_w = (const float*)d_in[1];
  const float* offset_b = (const float*)d_in[2];
  const float* doff_w   = (const float*)d_in[3];
  const float* doff_b   = (const float*)d_in[4];
  const float* dcn_w    = (const float*)d_in[5];
  const float* dcn_b    = (const float*)d_in[6];
  const float* sa_w     = (const float*)d_in[7];
  const float* sa_b     = (const float*)d_in[8];
  const float* ca_w     = (const float*)d_in[9];
  const float* ca_b     = (const float*)d_in[10];
  const float* gk_w     = (const float*)d_in[11];
  const float* gk_b     = (const float*)d_in[12];
  const float* fw[3]    = {(const float*)d_in[13], (const float*)d_in[15], (const float*)d_in[17]};
  const float* fb[3]    = {(const float*)d_in[14], (const float*)d_in[16], (const float*)d_in[18]};
  const float* fus_w    = (const float*)d_in[19];
  const float* fus_b    = (const float*)d_in[20];
  float* out = (float*)d_out;

  char* W = (char*)d_ws;
  size_t off = 0;
  auto take = [&](size_t bytes) -> void* {
    void* p = W + off;
    off = (off + bytes + 255) & ~(size_t)255;
    return p;
  };
  _Float16* colA  = (_Float16*)take((size_t)NIMG * HW_ * 576 * 2);
  _Float16* colB  = (_Float16*)take((size_t)NIMG * HW_ * 576 * 2); // also holds ykern f16 [img][576][HW]
  _Float16* wOff  = (_Float16*)take((size_t)32 * 576 * 2);
  _Float16* wDoff = (_Float16*)take((size_t)224 * 288 * 2);
  _Float16* wDcn  = (_Float16*)take((size_t)64 * 576 * 2);
  _Float16* wGk   = (_Float16*)take((size_t)576 * 64 * 2);
  _Float16* wF[3];
  for (int i = 0; i < 3; ++i) wF[i] = (_Float16*)take((size_t)64 * 576 * 2);
  _Float16* wFus  = (_Float16*)take((size_t)64 * 256 * 2);
  float* off_feat = (float*)take((size_t)NIMG * 32 * HW_ * 4);
  float* om       = (float*)take((size_t)NIMG * 216 * HW_ * 4);
  float* y1       = (float*)take((size_t)NIMG * 64 * HW_ * 4);
  float* tmpD     = (float*)take((size_t)NIMG * 64 * HW_ * 4);
  float* xd[3];
  for (int i = 0; i < 3; ++i) xd[i] = (float*)take((size_t)NIMG * 64 * HW_ * 4);
  float* meanY   = (float*)take((size_t)NIMG * 64 * 4);
  float* caB     = (float*)take((size_t)NIMG * 64 * 4);
  float* rowsum  = (float*)take((size_t)NIMG * 256 * H_ * 4);
  float* colsum  = (float*)take((size_t)NIMG * 256 * W_ * 4);
  float* chansum = (float*)take((size_t)NIMG * 256 * 4);
  float* shw     = (float*)take((size_t)NIMG * 256 * 4);
  float* scw     = (float*)take((size_t)NIMG * H_ * 4);
  float* sch     = (float*)take((size_t)NIMG * W_ * 4);

  auto cdiv = [](int a, int b) { return (a + b - 1) / b; };
  const int GX = HW_ / 256;   // pixel blocks: 4 waves x 64 pixels per block

  // 1. convert weights to f16 (padded)
  k_wcvt<<<cdiv(32 * 576, 256), 256, 0, stream>>>(offset_w, wOff, 32, 32, 576);
  k_wcvt<<<cdiv(224 * 288, 256), 256, 0, stream>>>(doff_w, wDoff, 216, 224, 288);
  k_wcvt<<<cdiv(64 * 576, 256), 256, 0, stream>>>(dcn_w, wDcn, 64, 64, 576);
  k_wcvt<<<cdiv(576 * 64, 256), 256, 0, stream>>>(gk_w, wGk, 576, 576, 64);
  for (int i = 0; i < 3; ++i)
    k_wcvt<<<cdiv(64 * 576, 256), 256, 0, stream>>>(fw[i], wF[i], 64, 64, 576);
  k_wcvt<<<cdiv(64 * 256, 256), 256, 0, stream>>>(fus_w, wFus, 64, 64, 256);

  // 2. offset conv: off_feat = lrelu(conv3x3(x))   [WMMA]
  k_im2col<<<cdiv(NIMG * HW_ * 64, 256), 256, 0, stream>>>(x, colA, 64);
  k_gemm<<<dim3(GX, 2, NIMG), 128, 0, stream>>>(wOff, colA, offset_b,
        off_feat, nullptr, nullptr, 32, 576, 1);

  // 3. dcn-offset conv: om = conv3x3(off_feat)     [WMMA]
  k_im2col<<<cdiv(NIMG * HW_ * 32, 256), 256, 0, stream>>>(off_feat, colB, 32);
  k_gemm<<<dim3(GX, 14, NIMG), 128, 0, stream>>>(wDoff, colB, doff_b,
        om, nullptr, nullptr, 216, 288, 0);

  // 4. modulated deformable sampling -> columns, then y1 = lrelu(GEMM) [WMMA]
  k_dcn_cols<<<cdiv(NIMG * HW_ * DG_, 256), 256, 0, stream>>>(x, om, colB);
  k_gemm<<<dim3(GX, 4, NIMG), 128, 0, stream>>>(wDcn, colB, dcn_b,
        y1, nullptr, nullptr, 64, 576, 1);

  // 5. SEKG -> y2 (f16 B operand in colA)
  k_chanmean<<<NIMG * 64, 256, 0, stream>>>(y1, meanY);
  k_ca<<<1, 128, 0, stream>>>(meanY, ca_w, ca_b, caB);
  k_sekg<<<cdiv(NIMG * 64 * HW_, 256), 256, 0, stream>>>(y1, sa_w, sa_b, caB, colA);

  // 6. dynamic kernels: ykern = 1x1 conv 64->576, f16 output into colB  [WMMA]
  k_gemm<<<dim3(GX, 36, NIMG), 128, 0, stream>>>(wGk, colA, gk_b,
        nullptr, colB, nullptr, 576, 64, 0);

  // 7. depth_dc at d = 1,3,5: dynamic depthwise -> im2col -> conv  [WMMA]
  const int dil[3] = {1, 3, 5};
  for (int i = 0; i < 3; ++i) {
    k_dyndw<<<cdiv(NIMG * 64 * HW_, 256), 256, 0, stream>>>(x, colB, tmpD, dil[i]);
    k_im2col<<<cdiv(NIMG * HW_ * 64, 256), 256, 0, stream>>>(tmpD, colA, 64);
    k_gemm<<<dim3(GX, 4, NIMG), 128, 0, stream>>>(wF[i], colA, fb[i],
          xd[i], nullptr, nullptr, 64, 576, 1);
  }

  // 8. attention over cat = [x, x1, x3, x5]
  k_catreduce<<<NIMG * 256, 128, 0, stream>>>(x, xd[0], xd[1], xd[2],
        rowsum, colsum, chansum);
  k_att<<<NIMG, 128, 0, stream>>>(rowsum, colsum, chansum, shw, scw, sch);
  k_fusb<<<cdiv(NIMG * HW_, 256), 256, 0, stream>>>(x, xd[0], xd[1], xd[2],
        shw, scw, sch, colA);

  // 9. fusion 1x1 conv 256->64 + bias + residual x0 -> d_out  [WMMA]
  k_gemm<<<dim3(GX, 4, NIMG), 128, 0, stream>>>(wFus, colA, fus_b,
        out, nullptr, x, 64, 256, 0);
}
